// GLFF_44375602102973
// MI455X (gfx1250) — matrix operand order repaired and checked by reading, not compiled, not run on hardware
//
#include <hip/hip_runtime.h>
#include <math.h>

// ---------------------------------------------------------------- types
typedef __attribute__((ext_vector_type(16))) __bf16          v16bf;
typedef __attribute__((ext_vector_type(8)))  __bf16          v8bf;
typedef __attribute__((ext_vector_type(8)))  float           v8f;
typedef __attribute__((ext_vector_type(8)))  unsigned short  u16x8;

#define DEV static __device__ __forceinline__

constexpr int Bn = 8, Cn = 128, Hn = 64, Wn = 64;
constexpr int HWn = Hn * Wn;          // 4096
constexpr int NBn = 256;              // 16x16 coarse blocks (S=4)
constexpr int C3n = 3 * Cn;           // 384
constexpr int KC_CONV = 36;           // (9*128)/32 k-chunks for conv
constexpr int KC_FIN  = 12;           // 384/32 k-chunks for final GEMM
constexpr int TROW = 144;             // padded LDS row (u16) for transpose tile

DEV unsigned short f2bf(float f) {                 // fp32 -> bf16 RNE
  unsigned int u = __builtin_bit_cast(unsigned int, f);
  unsigned int r = u + 0x7FFFu + ((u >> 16) & 1u);
  return (unsigned short)(r >> 16);
}

DEV v16bf load_frag16(const unsigned short* p) {   // 16 contiguous bf16
  u16x8 lo = *reinterpret_cast<const u16x8*>(p);
  u16x8 hi = *reinterpret_cast<const u16x8*>(p + 8);
  v8bf a = __builtin_bit_cast(v8bf, lo);
  v8bf b = __builtin_bit_cast(v8bf, hi);
  return __builtin_shufflevector(a, b, 0,1,2,3,4,5,6,7,8,9,10,11,12,13,14,15);
}

DEV int clampi(int v, int lo, int hi) { return v < lo ? lo : (v > hi ? hi : v); }

// ---------------------------------------------------------------- stage 1: build ins (F1 upsample, F2 copy, F3 pool)
__global__ void k_prep(const float* __restrict__ F1, const float* __restrict__ F2,
                       const float* __restrict__ F3, float* __restrict__ ins) {
  int t = blockIdx.x * 256 + threadIdx.x;
  int z = blockIdx.z;
  if (t >= Bn * Cn * HWn) return;
  int w = t & 63, h = (t >> 6) & 63, c = (t >> 12) & 127, b = t >> 19;
  float v;
  if (z == 0) {                                   // bilinear 32->64, half-pixel
    float fy = h * 0.5f - 0.25f, fx = w * 0.5f - 0.25f;
    int iy0 = (int)floorf(fy), ix0 = (int)floorf(fx);
    float ty = fy - (float)iy0, tx = fx - (float)ix0;
    int y0 = clampi(iy0, 0, 31), y1 = clampi(iy0 + 1, 0, 31);
    int x0 = clampi(ix0, 0, 31), x1 = clampi(ix0 + 1, 0, 31);
    const float* p = F1 + ((size_t)(b * Cn + c)) * 1024;
    float v00 = p[y0 * 32 + x0], v01 = p[y0 * 32 + x1];
    float v10 = p[y1 * 32 + x0], v11 = p[y1 * 32 + x1];
    v = (1.f - ty) * ((1.f - tx) * v00 + tx * v01) + ty * ((1.f - tx) * v10 + tx * v11);
  } else if (z == 1) {
    v = F2[((size_t)(b * Cn + c)) * HWn + h * Wn + w];
  } else {                                        // 2x2 mean pool 128->64
    const float* p = F3 + ((size_t)(b * Cn + c)) * 16384;
    int hh = 2 * h, ww = 2 * w;
    v = 0.25f * (p[hh * 128 + ww] + p[hh * 128 + ww + 1] +
                 p[(hh + 1) * 128 + ww] + p[(hh + 1) * 128 + ww + 1]);
  }
  ins[(((size_t)(b * 3 + z) * Cn + c) * HWn) + h * Wn + w] = v;
}

// ---------------------------------------------------------------- stage 1b: transpose ins -> pixel-major bf16 [bi][pix][c]
// LDS-tiled so both global reads (along w) and global writes (along c) are
// coalesced (reads f32, writes 16B vectors); fp32->bf16 conversion happens once here.
__global__ void __launch_bounds__(256) k_transp(const float* __restrict__ ins,
                                                unsigned short* __restrict__ insbf) {
  __shared__ alignas(16) unsigned short tl[64 * TROW];   // [pix64][c128 + pad]
  int bx = blockIdx.x;                                   // 24 * 64 tiles
  int pt = bx & 63, bi = bx >> 6;
  int pixbase = pt * 64;
  int tid = threadIdx.x;
  const float* src = ins + (size_t)bi * Cn * HWn + pixbase;
  // load: consecutive lanes -> consecutive pixels (coalesced f32)
  for (int idx = tid; idx < 64 * Cn; idx += 256) {
    int p = idx & 63, c = idx >> 6;
    tl[p * TROW + c] = f2bf(src[(size_t)c * HWn + p]);
  }
  __syncthreads();
  // store: consecutive lanes -> consecutive channels (coalesced 16B)
  unsigned short* dst = insbf + ((size_t)bi * HWn + pixbase) * Cn;
  for (int idx = tid; idx < 64 * 16; idx += 256) {       // 1024 u16x8 vectors
    int cg = idx & 15, p = idx >> 4;
    u16x8 v = *reinterpret_cast<const u16x8*>(&tl[p * TROW + cg * 8]);
    *reinterpret_cast<u16x8*>(dst + (size_t)p * Cn + cg * 8) = v;
  }
}

// ---------------------------------------------------------------- stage 2: 4x4 block sums of ins
__global__ void k_blocksum(const float* __restrict__ ins, float* __restrict__ Xsum) {
  int t = blockIdx.x * 256 + threadIdx.x;         // Bn*3*NBn*Cn
  if (t >= Bn * 3 * NBn * Cn) return;
  int c = t & 127, blk = (t >> 7) & 255, bi = t >> 15;
  int hb = blk >> 4, wb = blk & 15;
  const float* p = ins + ((size_t)bi * Cn + c) * HWn;
  float s = 0.f;
  for (int dy = 0; dy < 4; ++dy)
    for (int dx = 0; dx < 4; ++dx)
      s += p[(hb * 4 + dy) * Wn + wb * 4 + dx];
  Xsum[((size_t)bi * NBn + blk) * Cn + c] = s;
}

// ---------------------------------------------------------------- stage 3: block-level Q, K, Vb
__global__ void k_qkv(const float* __restrict__ Xsum,
                      const float* __restrict__ qw, const float* __restrict__ qb,
                      const float* __restrict__ kw, const float* __restrict__ kb,
                      const float* __restrict__ vw, const float* __restrict__ vb,
                      float* __restrict__ Q, float* __restrict__ K, float* __restrict__ Vb) {
  int t = blockIdx.x * 256 + threadIdx.x;         // Bn*3*NBn*Cn
  if (t >= Bn * 3 * NBn * Cn) return;
  int c = t & 127, n = (t >> 7) & 255, bi = t >> 15;
  int i = bi % 3;
  const float* x  = Xsum + ((size_t)bi * NBn + n) * Cn;
  const float* qr = qw + (size_t)(i * Cn + c) * Cn;
  const float* kr = kw + (size_t)(i * Cn + c) * Cn;
  const float* vr = vw + (size_t)(i * Cn + c) * Cn;
  float aq = 0.f, ak = 0.f, av = 0.f;
  for (int c2 = 0; c2 < Cn; ++c2) {
    float xv = x[c2];
    aq += xv * qr[c2]; ak += xv * kr[c2]; av += xv * vr[c2];
  }
  size_t idx = (size_t)t;
  Q[idx]  = aq * (1.f / 16.f) + qb[i * Cn + c];   // x_avg = Xsum/16
  K[idx]  = ak * (1.f / 16.f) + kb[i * Cn + c];
  Vb[idx] = av + 16.f * vb[i * Cn + c];           // blocksum(V) = Xsum@vw.T + 16*vb
}

// ---------------------------------------------------------------- stage 4: scores
__global__ void k_score(const float* __restrict__ Q, const float* __restrict__ K,
                        float* __restrict__ S) {
  int t = blockIdx.x * 256 + threadIdx.x;         // Bn*3*NBn*NBn
  if (t >= Bn * 3 * NBn * NBn) return;
  int m = t & 255, n = (t >> 8) & 255, bi = t >> 16;
  const float* q = Q + ((size_t)bi * NBn + n) * Cn;
  const float* k = K + ((size_t)bi * NBn + m) * Cn;
  float s = 0.f;
  for (int c = 0; c < Cn; ++c) s += q[c] * k[c];
  S[((size_t)bi * NBn + n) * NBn + m] = s * 0.08838834764831845f;   // 1/sqrt(128)
}

// ---------------------------------------------------------------- stage 5: row softmax (in-place)
__global__ void k_softmax(float* __restrict__ S) {
  int t = blockIdx.x * 256 + threadIdx.x;         // Bn*3*NBn rows
  if (t >= Bn * 3 * NBn) return;
  float* row = S + (size_t)t * NBn;
  float mx = -1e30f;
  for (int m = 0; m < NBn; ++m) mx = fmaxf(mx, row[m]);
  float sum = 0.f;
  for (int m = 0; m < NBn; ++m) { float e = __expf(row[m] - mx); row[m] = e; sum += e; }
  float inv = 1.f / sum;
  for (int m = 0; m < NBn; ++m) row[m] *= inv;
}

// ---------------------------------------------------------------- stage 6: ob = A @ Vb
__global__ void k_ob(const float* __restrict__ S, const float* __restrict__ Vb,
                     float* __restrict__ ob) {
  int t = blockIdx.x * 256 + threadIdx.x;         // Bn*3*NBn*Cn
  if (t >= Bn * 3 * NBn * Cn) return;
  int c = t & 127, n = (t >> 7) & 255, bi = t >> 15;
  const float* srow = S + ((size_t)bi * NBn + n) * NBn;
  const float* v = Vb + (size_t)bi * NBn * Cn + c;
  float a = 0.f;
  for (int m = 0; m < NBn; ++m) a += srow[m] * v[(size_t)m * Cn];
  ob[(size_t)t] = a;
}

// ---------------------------------------------------------------- stage 7: pack conv weights into WMMA A-fragments
// k-order: k = (ky*3+kx)*128 + c ; chunk kc: spatial s = kc/4, c-chunk = kc%4
// A-frag element e of lane (g = lane/16): K = g*16 + e
__global__ void k_packw(const float* __restrict__ conv_w, unsigned short* __restrict__ pw) {
  int t = blockIdx.x * 256 + threadIdx.x;         // 3*36*8*32
  if (t >= 3 * KC_CONV * 8 * 32) return;
  int lane = t & 31, ot = (t >> 5) & 7;
  int kc = (t >> 8) % KC_CONV, i = (t >> 8) / KC_CONV;
  int g = lane >> 4, mrow = lane & 15;
  int s = kc >> 2, cc = kc & 3;
  int ky = s / 3, kx = s % 3;
  int o = ot * 16 + mrow;
  unsigned short* dst = pw + (size_t)t * 16;
#pragma unroll
  for (int e = 0; e < 16; ++e) {
    int c = cc * 32 + g * 16 + e;
    float wv = conv_w[((((size_t)i * Cn + o) * Cn + c) * 3 + ky) * 3 + kx];
    dst[e] = f2bf(wv);
  }
}

__global__ void k_zero(float* __restrict__ p, int n) {
  int t = blockIdx.x * 256 + threadIdx.x;
  if (t < n) p[t] = 0.f;
}

// ---------------------------------------------------------------- stage 8: conv3x3 (implicit GEMM, WMMA bf16)
// block = (b, scale, h, wtile16): 8 waves x (16 o x 16 px) = 128 o x 16 px
__global__ void __launch_bounds__(256) k_conv(const unsigned short* __restrict__ insbf,
                                              const unsigned short* __restrict__ pw,
                                              const float* __restrict__ conv_b,
                                              const float* __restrict__ ob,
                                              unsigned short* __restrict__ catbf,
                                              float* __restrict__ chmean) {
  __shared__ alignas(16) unsigned short lds[3 * 18 * 128];   // [row3][col18][c128] bf16
  int bx = blockIdx.x;
  int wt = bx & 3, h = (bx >> 2) & 63;
  int bi = bx >> 8, i = bi % 3, b = bi / 3;
  int wbase = wt * 16;
  int tid = threadIdx.x;

  // halo patch fill: contiguous u16x8 copies from pixel-major bf16 tensor
  const unsigned short* srcb = insbf + (size_t)bi * HWn * Cn;
  for (int idx = tid; idx < 3 * 18 * 16; idx += 256) {       // 864 vectors
    int c8  = idx & 15;
    int col = (idx >> 4) % 18;
    int row = (idx >> 4) / 18;
    int hh = h + row - 1, ww = wbase + col - 1;
    u16x8 v = {0, 0, 0, 0, 0, 0, 0, 0};
    if (hh >= 0 && hh < Hn && ww >= 0 && ww < Wn)
      v = *reinterpret_cast<const u16x8*>(srcb + ((size_t)(hh * Wn + ww)) * Cn + c8 * 8);
    *reinterpret_cast<u16x8*>(&lds[(row * 18 + col) * 128 + c8 * 8]) = v;
  }
  __syncthreads();

  int wave = tid >> 5, lane = tid & 31, g = lane >> 4, n = lane & 15;
  v8f acc = {0.f, 0.f, 0.f, 0.f, 0.f, 0.f, 0.f, 0.f};
  const unsigned short* pwi = pw + (size_t)i * KC_CONV * 8 * 32 * 16;
  __builtin_prefetch(pwi + (((size_t)wave * 32) + lane) * 16, 0, 3);

  for (int kc = 0; kc < KC_CONV; ++kc) {
    v16bf a = load_frag16(pwi + (((size_t)kc * 8 + wave) * 32 + lane) * 16);
    int s = kc >> 2, cc = kc & 3;
    int ky = s / 3, kx = s % 3;
    // B-frag: K = g*16 + e -> channel c = cc*32 + g*16 + e (contiguous in LDS)
    int off = (ky * 18 + (n + kx)) * 128 + cc * 32 + g * 16;
    v16bf bb = load_frag16(&lds[off]);
    acc = __builtin_amdgcn_wmma_f32_16x16x32_bf16(false, a, false, bb,
                                                  (short)0, acc, false, false);
  }

  int obase = wave * 16;
  int pix = h * Wn + wbase + n;
  int blk = (h >> 2) * 16 + ((wbase + n) >> 2);
  const float* obp = ob + ((size_t)bi * NBn + blk) * Cn;

  u16x8 st;
  float vals[8];
#pragma unroll
  for (int r = 0; r < 8; ++r) {                    // D vgpr r -> o = obase + r + 8g
    int o = obase + r + 8 * g;
    float v = acc[r] + conv_b[i * Cn + o] + obp[o];
    vals[r] = v;
    st[r] = f2bf(v);
  }
  // channels obase+8g .. +7 are contiguous -> one aligned 16B store
  *reinterpret_cast<u16x8*>(catbf + ((size_t)(b * HWn + pix)) * C3n + i * Cn + obase + 8 * g) = st;

#pragma unroll
  for (int r = 0; r < 8; ++r) {                    // channel-mean partials for SE
    float red = vals[r];
    for (int msk = 1; msk < 16; msk <<= 1) red += __shfl_xor(red, msk, 16);
    if (n == 0) {
      int o = obase + r + 8 * g;
      atomicAdd(&chmean[b * C3n + i * Cn + o], red);
    }
  }
}

// ---------------------------------------------------------------- stage 9: U = out_w @ c3d_w (batch-independent fold)
__global__ void k_foldU(const float* __restrict__ out_w, const float* __restrict__ c3d_w,
                        const float* __restrict__ c3d_b, const float* __restrict__ out_b,
                        float* __restrict__ U, float* __restrict__ bias2) {
  int t = blockIdx.x * 256 + threadIdx.x;          // Cn*C3n
  if (t >= Cn * C3n) return;
  int k = t % C3n, o2 = t / C3n;
  int c = k & 127, d = k >> 7;                     // cat channel k = d*128 + c
  float s = 0.f;
  for (int o = 0; o < Cn; ++o) s += out_w[o2 * Cn + o] * c3d_w[((size_t)o * Cn + c) * 3 + d];
  U[t] = s;
  if (k == 0) {
    float bsum = out_b[o2];
    for (int o = 0; o < Cn; ++o) bsum += out_w[o2 * Cn + o] * c3d_b[o];
    bias2[o2] = bsum;
  }
}

// ---------------------------------------------------------------- stage 10: SE MLP -> y
__global__ void k_se(const float* __restrict__ chmean,
                     const float* __restrict__ ca1_w, const float* __restrict__ ca1_b,
                     const float* __restrict__ ca2_w, const float* __restrict__ ca2_b,
                     float* __restrict__ yse) {
  int b = blockIdx.x, tid = threadIdx.x;
  __shared__ float y0[C3n];
  __shared__ float hid[24];
  for (int ch = tid; ch < C3n; ch += 256) y0[ch] = chmean[b * C3n + ch] * (1.f / 4096.f);
  __syncthreads();
  if (tid < 24) {
    float s = ca1_b[tid];
    for (int ch = 0; ch < C3n; ++ch) s += ca1_w[tid * C3n + ch] * y0[ch];
    hid[tid] = fmaxf(s, 0.f);
  }
  __syncthreads();
  for (int ch = tid; ch < C3n; ch += 256) {
    float s = ca2_b[ch];
    for (int j = 0; j < 24; ++j) s += ca2_w[ch * 24 + j] * hid[j];
    yse[b * C3n + ch] = 1.f / (1.f + __expf(-s));
  }
}

// ---------------------------------------------------------------- stage 11: pack W2 = U * y[b] into A-fragments
__global__ void k_packW2(const float* __restrict__ U, const float* __restrict__ yse,
                         unsigned short* __restrict__ pw2) {
  int t = blockIdx.x * 256 + threadIdx.x;          // Bn*12*8*32
  if (t >= Bn * KC_FIN * 8 * 32) return;
  int lane = t & 31, ot = (t >> 5) & 7;
  int kc = (t >> 8) % KC_FIN, b = (t >> 8) / KC_FIN;
  int g = lane >> 4, mrow = lane & 15;
  int o2 = ot * 16 + mrow;
  unsigned short* dst = pw2 + (size_t)t * 16;
#pragma unroll
  for (int e = 0; e < 16; ++e) {
    int k = kc * 32 + g * 16 + e;
    dst[e] = f2bf(U[o2 * C3n + k] * yse[b * C3n + k]);
  }
}

// ---------------------------------------------------------------- stage 12: final fused GEMM (128x384)@(384x4096) per batch
__global__ void __launch_bounds__(256) k_final(const unsigned short* __restrict__ pw2,
                                               const unsigned short* __restrict__ catbf,
                                               const float* __restrict__ bias2,
                                               float* __restrict__ out) {
  int bx = blockIdx.x;                             // Bn * 256 pixel-tiles
  int pt = bx & 255, b = bx >> 8;
  int tid = threadIdx.x, wave = tid >> 5, lane = tid & 31, g = lane >> 4, n = lane & 15;
  int pix = pt * 16 + n;
  v8f acc = {0.f, 0.f, 0.f, 0.f, 0.f, 0.f, 0.f, 0.f};
  const unsigned short* pa = pw2 + (size_t)b * KC_FIN * 8 * 32 * 16;
  const unsigned short* pb = catbf + ((size_t)(b * HWn + pix)) * C3n + g * 16;
  __builtin_prefetch(pb, 0, 3);                    // global_prefetch for the cat row
  for (int kc = 0; kc < KC_FIN; ++kc) {
    v16bf a  = load_frag16(pa + (((size_t)kc * 8 + wave) * 32 + lane) * 16);
    v16bf bb = load_frag16(pb + kc * 32);          // ch = kc*32 + g*16 + e, contiguous
    acc = __builtin_amdgcn_wmma_f32_16x16x32_bf16(false, a, false, bb,
                                                  (short)0, acc, false, false);
  }
  int obase = wave * 16;
#pragma unroll
  for (int r = 0; r < 8; ++r) {
    int o = obase + r + 8 * g;
    out[((size_t)b * Cn + o) * HWn + pix] = acc[r] + bias2[o];
  }
}

// ---------------------------------------------------------------- launcher
extern "C" void kernel_launch(void* const* d_in, const int* in_sizes, int n_in,
                              void* d_out, int out_size, void* d_ws, size_t ws_size,
                              hipStream_t stream) {
  (void)in_sizes; (void)n_in; (void)out_size; (void)ws_size;
  const float* F1     = (const float*)d_in[0];
  const float* F2     = (const float*)d_in[1];
  const float* F3     = (const float*)d_in[2];
  const float* conv_w = (const float*)d_in[3];
  const float* conv_b = (const float*)d_in[4];
  const float* qw     = (const float*)d_in[5];
  const float* qb     = (const float*)d_in[6];
  const float* kw     = (const float*)d_in[7];
  const float* kb     = (const float*)d_in[8];
  const float* vw     = (const float*)d_in[9];
  const float* vb     = (const float*)d_in[10];
  const float* ca1_w  = (const float*)d_in[11];
  const float* ca1_b  = (const float*)d_in[12];
  const float* ca2_w  = (const float*)d_in[13];
  const float* ca2_b  = (const float*)d_in[14];
  const float* c3d_w  = (const float*)d_in[15];
  const float* c3d_b  = (const float*)d_in[16];
  const float* out_w  = (const float*)d_in[17];
  const float* out_b  = (const float*)d_in[18];
  float* outp = (float*)d_out;

  // workspace layout (all sizes multiples of 256B)
  char* base = (char*)d_ws;
  size_t off = 0;
  auto take = [&](size_t bytes) { char* p = base + off; off += bytes; return p; };
  float*          ins    = (float*)         take((size_t)3 * Bn * Cn * HWn * 4);
  unsigned short* insbf  = (unsigned short*)take((size_t)3 * Bn * HWn * Cn * 2);
  float*          Xsum   = (float*)         take((size_t)Bn * 3 * NBn * Cn * 4);
  float*          Qb     = (float*)         take((size_t)Bn * 3 * NBn * Cn * 4);
  float*          Kb     = (float*)         take((size_t)Bn * 3 * NBn * Cn * 4);
  float*          Vbb    = (float*)         take((size_t)Bn * 3 * NBn * Cn * 4);
  float*          Sc     = (float*)         take((size_t)Bn * 3 * NBn * NBn * 4);
  float*          obuf   = (float*)         take((size_t)Bn * 3 * NBn * Cn * 4);
  unsigned short* pw     = (unsigned short*)take((size_t)3 * KC_CONV * 8 * 32 * 16 * 2);
  unsigned short* catbf  = (unsigned short*)take((size_t)Bn * HWn * C3n * 2);
  float*          chmean = (float*)         take((size_t)Bn * C3n * 4);
  float*          U      = (float*)         take((size_t)Cn * C3n * 4);
  float*          bias2  = (float*)         take((size_t)Cn * 4);
  float*          yse    = (float*)         take((size_t)Bn * C3n * 4);
  unsigned short* pw2    = (unsigned short*)take((size_t)Bn * KC_FIN * 8 * 32 * 16 * 2);

  dim3 gprep((Bn * Cn * HWn) / 256, 1, 3);
  k_prep<<<gprep, 256, 0, stream>>>(F1, F2, F3, ins);
  k_transp<<<Bn * 3 * (HWn / 64), 256, 0, stream>>>(ins, insbf);
  k_blocksum<<<(Bn * 3 * NBn * Cn) / 256, 256, 0, stream>>>(ins, Xsum);
  k_qkv<<<(Bn * 3 * NBn * Cn) / 256, 256, 0, stream>>>(Xsum, qw, qb, kw, kb, vw, vb, Qb, Kb, Vbb);
  k_score<<<(Bn * 3 * NBn * NBn) / 256, 256, 0, stream>>>(Qb, Kb, Sc);
  k_softmax<<<(Bn * 3 * NBn) / 256, 256, 0, stream>>>(Sc);
  k_ob<<<(Bn * 3 * NBn * Cn) / 256, 256, 0, stream>>>(Sc, Vbb, obuf);
  k_packw<<<(3 * KC_CONV * 8 * 32) / 256, 256, 0, stream>>>(conv_w, pw);
  k_zero<<<(Bn * C3n + 255) / 256, 256, 0, stream>>>(chmean, Bn * C3n);
  k_conv<<<Bn * 3 * Hn * 4, 256, 0, stream>>>(insbf, pw, conv_b, obuf, catbf, chmean);
  k_foldU<<<(Cn * C3n) / 256, 256, 0, stream>>>(out_w, c3d_w, c3d_b, out_b, U, bias2);
  k_se<<<Bn, 256, 0, stream>>>(chmean, ca1_w, ca1_b, ca2_w, ca2_b, yse);
  k_packW2<<<(Bn * KC_FIN * 8 * 32) / 256, 256, 0, stream>>>(U, yse, pw2);
  k_final<<<Bn * 256, 256, 0, stream>>>(pw2, catbf, bias2, outp);
}
